// SegTransformerDecoder_3006477107639
// MI455X (gfx1250) — compile-verified, matrix operand order checked by reading, not run on hardware
//
#include <hip/hip_runtime.h>

#define ACT_NONE 0
#define ACT_GELU 1
#define ACT_RELU 2

typedef __attribute__((ext_vector_type(16))) __bf16 bfvec16;
typedef __attribute__((ext_vector_type(8)))  float  fvec8;

__device__ __forceinline__ unsigned short f2bf(float f) {
  unsigned u = __float_as_uint(f);
  unsigned r = u + 0x7FFFu + ((u >> 16) & 1u);   // round-to-nearest-even
  return (unsigned short)(r >> 16);
}

// ---------------------------------------------------------------------------
// Generic WMMA GEMM / implicit-GEMM conv:
//   D(Cout x HW) = act( A(Cout x KK) * B(KK x HW) + bias ) + res
// A  : weights, element (m,k) at Aw[m*aSM + k*aSK]   (OIHW 1x1/3x3, or W^T)
// B  : k3==1 -> X[k*HW + n] ; k3==3 -> zero-padded 3x3 im2col gather from NCHW
// Block: 256 threads = 8 waves (4M x 2N), block tile 128(M) x 64(N).
// Wave tile 32x32 = 2x2 WMMA subtiles -> 4 chained v_wmma_f32_16x16x32_bf16
// per 32-wide K step, with 2x fragment reuse on both A and B.
// ---------------------------------------------------------------------------
__global__ __launch_bounds__(256) void gemm_conv_wmma(
    const float* __restrict__ Aw, const float* __restrict__ X,
    const float* __restrict__ bias, const float* __restrict__ res,
    float* __restrict__ D,
    int Cout, int HW, int K0, int H, int W, int k3,
    int aSM, int aSK, int act)
{
  __shared__ unsigned short lsA[8][1024];  // per-wave 32(m) x 32(k) bf16
  __shared__ unsigned short lsB[8][1024];  // per-wave 32(n) x 32(k) bf16 (transposed)
  const int wid  = threadIdx.x >> 5;
  const int lane = threadIdx.x & 31;
  const int m0 = blockIdx.y * 128 + (wid >> 1) * 32;
  const int n0 = blockIdx.x * 64  + (wid & 1) * 32;
  const int KK = K0 * k3 * k3;
  const int lm = lane & 15;
  const int kb = (lane >> 4) << 3;         // 0 or 8 : K sub-block AND M/N row offset
  unsigned short* sA = lsA[wid];
  unsigned short* sB = lsB[wid];
  fvec8 acc00 = {0.f,0.f,0.f,0.f,0.f,0.f,0.f,0.f};
  fvec8 acc01 = acc00, acc10 = acc00, acc11 = acc00;

  for (int k0 = 0; k0 < KK; k0 += 32) {
    // prefetch next A K-slice while we consume this one (global_prefetch_b8)
    if (k0 + 32 < KK) {
      int pm = m0 + lane; if (pm >= Cout) pm = m0;
      __builtin_prefetch(&Aw[pm * aSM + (k0 + 32) * aSK], 0, 1);
    }
    // ---- stage A tile: rows m0..m0+31, k0..k0+31, LDS layout [m][k] ----
#pragma unroll 4
    for (int i = 0; i < 32; ++i) {
      int m = m0 + i, k = k0 + lane;
      float v = 0.0f;
      if (m < Cout && k < KK) v = Aw[m * aSM + k * aSK];
      sA[i * 32 + lane] = f2bf(v);
    }
    // ---- stage B tile (k-major gather, stored transposed [n][k]) ----
#pragma unroll 4
    for (int i = 0; i < 32; ++i) {
      int kl = ((i & 15) << 1) + (lane >> 4);
      int nl = ((i >> 4) << 4) + lm;
      int kk = k0 + kl;
      int n  = n0 + nl;
      float v = 0.0f;
      if (kk < KK && n < HW) {
        if (k3 == 1) {
          v = X[kk * HW + n];
        } else {
          int cin = kk / 9;
          int r   = kk - cin * 9;
          int y = n / W + r / 3 - 1;
          int x = n % W + (r % 3) - 1;
          if (y >= 0 && y < H && x >= 0 && x < W) v = X[(cin * H + y) * W + x];
        }
      }
      sB[nl * 32 + kl] = f2bf(v);
    }
    __builtin_amdgcn_wave_barrier();
    // ---- build WMMA fragments per ISA 16-bit A/B layout ----
    auto loadFrag = [&](const unsigned short* s, int row) -> bfvec16 {
      union { bfvec16 v; unsigned u[8]; } f;
      const int base = row * 32 + kb;
#pragma unroll
      for (int t = 0; t < 4; ++t) {
        __builtin_memcpy(&f.u[t],     &s[base + 2 * t],      4);
        __builtin_memcpy(&f.u[4 + t], &s[base + 16 + 2 * t], 4);
      }
      return f.v;
    };
    bfvec16 a0 = loadFrag(sA, lm);
    bfvec16 a1 = loadFrag(sA, lm + 16);
    bfvec16 b0 = loadFrag(sB, lm);
    bfvec16 b1 = loadFrag(sB, lm + 16);
    acc00 = __builtin_amdgcn_wmma_f32_16x16x32_bf16(false, a0, false, b0, (short)0, acc00, false, false);
    acc01 = __builtin_amdgcn_wmma_f32_16x16x32_bf16(false, a0, false, b1, (short)0, acc01, false, false);
    acc10 = __builtin_amdgcn_wmma_f32_16x16x32_bf16(false, a1, false, b0, (short)0, acc10, false, false);
    acc11 = __builtin_amdgcn_wmma_f32_16x16x32_bf16(false, a1, false, b1, (short)0, acc11, false, false);
    __builtin_amdgcn_wave_barrier();
  }
  // ---- epilogue: C/D layout lane=(N), elem r -> M = r + 8*(lane/16) ----
  auto emit = [&](fvec8& a, int smOff, int snOff) {
#pragma unroll
    for (int r = 0; r < 8; ++r) {
      int mm = m0 + smOff + kb + r;
      int nn = n0 + snOff + lm;
      if (mm < Cout && nn < HW) {
        float v = a[r];
        if (bias) v += bias[mm];
        if (act == ACT_GELU)      v = 0.5f * v * (1.0f + erff(v * 0.70710678118654752f));
        else if (act == ACT_RELU) v = fmaxf(v, 0.0f);
        if (res) v += res[mm * HW + nn];
        D[mm * HW + nn] = v;
      }
    }
  };
  emit(acc00, 0, 0);
  emit(acc01, 0, 16);
  emit(acc10, 16, 0);
  emit(acc11, 16, 16);
}

// ---------------------------------------------------------------------------
// Instance / batch norm over HW per channel (B=1 so they coincide), optional
// affine (g,b) and fused ReLU. One block per channel.
// ---------------------------------------------------------------------------
__global__ __launch_bounds__(256) void norm_kernel(
    float* __restrict__ x, const float* __restrict__ g,
    const float* __restrict__ b, int HW, int relu)
{
  __shared__ float red0[256];
  __shared__ float red1[256];
  const int c = blockIdx.x;
  float s = 0.f, s2 = 0.f;
  for (int i = threadIdx.x; i < HW; i += 256) {
    float v = x[(long)c * HW + i];
    s += v; s2 += v * v;
  }
  red0[threadIdx.x] = s; red1[threadIdx.x] = s2;
  __syncthreads();
  for (int o = 128; o > 0; o >>= 1) {
    if (threadIdx.x < o) {
      red0[threadIdx.x] += red0[threadIdx.x + o];
      red1[threadIdx.x] += red1[threadIdx.x + o];
    }
    __syncthreads();
  }
  float mean = red0[0] / HW;
  float var  = red1[0] / HW - mean * mean;
  float rs   = rsqrtf(var + 1e-5f);
  float ga = g ? g[c] : 1.f;
  float be = b ? b[c] : 0.f;
  for (int i = threadIdx.x; i < HW; i += 256) {
    float v = (x[(long)c * HW + i] - mean) * rs * ga + be;
    if (relu) v = fmaxf(v, 0.f);
    x[(long)c * HW + i] = v;
  }
}

// half-pixel bilinear resize; chlast=1 -> HWC layout, else CHW
__global__ void resize_bilinear_k(const float* __restrict__ in, float* __restrict__ out,
                                  int C, int IH, int IW, int OH, int OW, int chlast)
{
  int idx = blockIdx.x * blockDim.x + threadIdx.x;
  if (idx >= C * OH * OW) return;
  int c, oy, ox;
  if (chlast) { c = idx % C; int t = idx / C; ox = t % OW; oy = t / OW; }
  else        { ox = idx % OW; int t = idx / OW; oy = t % OH; c = t / OH; }
  float cy = fminf(fmaxf((oy + 0.5f) * (float)IH / OH - 0.5f, 0.f), IH - 1.f);
  float cx = fminf(fmaxf((ox + 0.5f) * (float)IW / OW - 0.5f, 0.f), IW - 1.f);
  int y0 = (int)floorf(cy); int y1 = min(y0 + 1, IH - 1); float ty = cy - y0;
  int x0 = (int)floorf(cx); int x1 = min(x0 + 1, IW - 1); float tx = cx - x0;
  #define AT(y, xx) (chlast ? in[((long)(y) * IW + (xx)) * C + c] : in[((long)c * IH + (y)) * IW + (xx)])
  float top = AT(y0, x0) * (1.f - tx) + AT(y0, x1) * tx;
  float bot = AT(y1, x0) * (1.f - tx) + AT(y1, x1) * tx;
  #undef AT
  float v = top * (1.f - ty) + bot * ty;
  if (chlast) out[((long)oy * OW + ox) * C + c] = v;
  else        out[((long)c * OH + oy) * OW + ox] = v;
}

__global__ void resize_nearest_k(const float* __restrict__ in, float* __restrict__ out,
                                 int C, int IH, int IW, int OH, int OW)
{
  int idx = blockIdx.x * blockDim.x + threadIdx.x;
  if (idx >= C * OH * OW) return;
  int ox = idx % OW; int t = idx / OW; int oy = t % OH; int c = t / OH;
  int yi = oy * IH / OH; int xi = ox * IW / OW;
  out[((long)c * OH + oy) * OW + ox] = in[((long)c * IH + yi) * IW + xi];
}

__global__ void add_relu_k(const float* __restrict__ a, const float* __restrict__ b,
                           float* __restrict__ o, int n)
{
  int i = blockIdx.x * blockDim.x + threadIdx.x;
  if (i < n) o[i] = fmaxf(a[i] + b[i], 0.f);
}

// ---------------------------------------------------------------------------
// Deformable sampler: per (pixel q, point p) compute reference point + offset,
// project through 6 cameras, softmax level weights, 4-level bilinear taps.
// One thread per channel c (128). Also emits normalized pts (3 x 4*hw) for the
// pos-MLP GEMMs. sampled layout: channel (p*128+c), pixel-major.
// ---------------------------------------------------------------------------
__global__ __launch_bounds__(128) void sample_kernel(
    const float* __restrict__ offc, const float* __restrict__ swc,
    const float* __restrict__ ref,  const float* __restrict__ l2i,
    const float* __restrict__ f0, const float* __restrict__ f1,
    const float* __restrict__ f2, const float* __restrict__ f3,
    float* __restrict__ sampled, float* __restrict__ rpn_t,
    int hw, float scale)
{
  const int q = blockIdx.x;
  const int p = blockIdx.y;
  const int c = threadIdx.x;
  const float LO[3]   = {-51.2f, -51.2f, -5.0f};
  const float SPAN[3] = {102.4f, 102.4f,  8.0f};
  const float axy = 0.25f * scale + 1e-6f;
  const float amp[3] = {axy, axy, 4.0f + 1e-6f};
  float rp[3];
#pragma unroll
  for (int j = 0; j < 3; ++j) {
    float o = offc[(long)(p * 3 + j) * hw + q];
    o = 1.f / (1.f + expf(-o));
    o = o * 2.f * amp[j] - amp[j];
    rp[j] = ref[(long)q * 3 + j] * SPAN[j] + LO[j] + o;
  }
  if (c < 3) rpn_t[(long)c * (hw * 4) + (long)p * hw + q] = (rp[c] - LO[c]) / SPAN[c];

  float lg[4], mx = -1e30f;
#pragma unroll
  for (int l = 0; l < 4; ++l) { lg[l] = swc[(long)(p * 4 + l) * hw + q]; mx = fmaxf(mx, lg[l]); }
  float se = 0.f;
#pragma unroll
  for (int l = 0; l < 4; ++l) { lg[l] = expf(lg[l] - mx); se += lg[l]; }
#pragma unroll
  for (int l = 0; l < 4; ++l) lg[l] /= se;

  const int HL[4] = {32, 16, 8, 4};
  const int WL[4] = {88, 44, 22, 11};
  const float* FP[4] = {f0, f1, f2, f3};
  float acc = 0.f;
  for (int n = 0; n < 6; ++n) {
    const float* M = l2i + n * 16;
    float c0 = M[0] * rp[0] + M[1] * rp[1] + M[2]  * rp[2] + M[3];
    float c1 = M[4] * rp[0] + M[5] * rp[1] + M[6]  * rp[2] + M[7];
    float c2 = M[8] * rp[0] + M[9] * rp[1] + M[10] * rp[2] + M[11];
    float denom = fmaxf(c2, 1e-5f);
    float u = c0 / denom / 704.f;
    float v = c1 / denom / 256.f;
    bool valid = (c2 > 1e-5f) && (u >= 0.f) && (u <= 1.f) && (v >= 0.f) && (v <= 1.f);
    if (!valid) continue;
    float sacc = 0.f;
    for (int l = 0; l < 4; ++l) {
      int Hl = HL[l], Wl = WL[l];
      float px = u * Wl - 0.5f, py = v * Hl - 0.5f;
      float x0f = floorf(px), y0f = floorf(py);
      int x0 = (int)x0f, y0 = (int)y0f;
      float tx = px - x0f, ty = py - y0f;
      const float* Fb = FP[l] + (long)(n * 128 + c) * Hl * Wl;
      #define TAP(yy, xx) (((yy) < 0 || (yy) >= Hl || (xx) < 0 || (xx) >= Wl) ? 0.f : Fb[(yy) * Wl + (xx)])
      float sv = TAP(y0, x0)     * (1.f - ty) * (1.f - tx)
               + TAP(y0, x0 + 1) * (1.f - ty) * tx
               + TAP(y0 + 1, x0) * ty * (1.f - tx)
               + TAP(y0 + 1, x0 + 1) * ty * tx;
      #undef TAP
      sacc += sv * lg[l];
    }
    acc += sacc;
  }
  sampled[(long)(p * 128 + c) * hw + q] = acc;
}

// sampled[(p*128+c)*hw + q] += pos[(c*4 + p)*hw + q]
__global__ void add_pos_k(float* __restrict__ sampled, const float* __restrict__ pos, int hw)
{
  long i = (long)blockIdx.x * blockDim.x + threadIdx.x;
  if (i >= (long)512 * hw) return;
  long q = i % hw; long ch = i / hw;
  int p = (int)(ch >> 7); int c = (int)(ch & 127);
  sampled[i] += pos[((long)c * 4 + p) * hw + q];
}

// ---------------------------------------------------------------------------
extern "C" void kernel_launch(void* const* d_in, const int* in_sizes, int n_in,
                              void* d_out, int out_size, void* d_ws, size_t ws_size,
                              hipStream_t stream)
{
  (void)in_sizes; (void)out_size; (void)ws_size;
  if (n_in < 63) return;
  auto F = [&](int i) -> const float* { return (const float*)d_in[i]; };
  const float *f0 = F(0), *f1 = F(1), *f2 = F(2), *f3 = F(3);
  const float *l2i = F(4), *bevq = F(5), *bevp = F(6);
  // params pytree order: dec{bn1_b,bn1_g,bn2_b,bn2_g,c1_w,c2_w,up_b,up_w},
  // layers[0..1]{in_conv(w,b)x3, mid_conv(w,b)x3, out_conv(w,b)x3, so_b,so_w,sw_b,sw_w},
  // pos{b1,b2,w1,w2}
  const float *bn1_b = F(7), *bn1_g = F(8), *bn2_b = F(9), *bn2_g = F(10);
  const float *dc1 = F(11), *dc2 = F(12), *up_b = F(13), *up_w = F(14);
  struct LP { const float *iw0,*ib0,*iw1,*ib1,*iw2,*ib2,
                          *mw0,*mb0,*mw1,*mb1,*mw2,*mb2,
                          *ow0,*ob0,*ow1,*ob1,*ow2,*ob2,
                          *so_b,*so_w,*sw_b,*sw_w; };
  auto loadLP = [&](int b) { LP L;
    L.iw0=F(b+0);  L.ib0=F(b+1);  L.iw1=F(b+2);  L.ib1=F(b+3);  L.iw2=F(b+4);  L.ib2=F(b+5);
    L.mw0=F(b+6);  L.mb0=F(b+7);  L.mw1=F(b+8);  L.mb1=F(b+9);  L.mw2=F(b+10); L.mb2=F(b+11);
    L.ow0=F(b+12); L.ob0=F(b+13); L.ow1=F(b+14); L.ob1=F(b+15); L.ow2=F(b+16); L.ob2=F(b+17);
    L.so_b=F(b+18); L.so_w=F(b+19); L.sw_b=F(b+20); L.sw_w=F(b+21);
    return L; };
  LP L0 = loadLP(15), L1 = loadLP(37);
  const float *pb1 = F(59), *pb2 = F(60), *pw1 = F(61), *pw2 = F(62);

  // workspace bump arena
  char* base = (char*)d_ws;
  size_t off = 0;
  auto alloc = [&](size_t nflt) -> float* {
    float* p = (float*)(base + off);
    off += ((nflt * sizeof(float)) + 255) & ~(size_t)255;
    return p;
  };
  auto gemm = [&](const float* A, const float* X, const float* bias, const float* res,
                  float* D, int Cout, int HW, int K0, int H, int W, int k3,
                  int aSM, int aSK, int act) {
    dim3 g((HW + 63) / 64, (Cout + 127) / 128);
    gemm_conv_wmma<<<g, 256, 0, stream>>>(A, X, bias, res, D, Cout, HW, K0, H, W, k3, aSM, aSK, act);
  };
  auto norm = [&](float* x, const float* g, const float* b, int C, int HW, int relu) {
    norm_kernel<<<C, 256, 0, stream>>>(x, g, b, HW, relu);
  };
  auto ew = [&](int n) { return dim3((n + 255) / 256); };

  float* q3   = alloc(512 * 625);     // layer-1 output (persists through decoder)
  float* bev2 = alloc(128 * 10000);   // decoder output (persists through layer-2)
  const size_t mark = off;

  // ================= Layer 1: d=512, 25x25, scale=4 =================
  {
    off = mark;
    const int hw = 625, H = 25, W = 25, d = 512;
    float* ref25 = alloc(hw * 3);
    resize_bilinear_k<<<ew(3 * hw), 256, 0, stream>>>(bevp, ref25, 3, 100, 100, 25, 25, 1);
    float* t1 = alloc(d * hw); float* t2 = alloc(d * hw); float* q = alloc(d * hw);
    gemm(L0.iw0, bevq, L0.ib0, nullptr, t1, d, hw, d, H, W, 1, d, 1, ACT_GELU);
    gemm(L0.iw1, t1,   L0.ib1, nullptr, t2, d, hw, d, H, W, 3, d * 9, 1, ACT_GELU);
    gemm(L0.iw2, t2,   L0.ib2, bevq,    q,  d, hw, d, H, W, 1, d, 1, ACT_NONE);
    norm(q, nullptr, nullptr, d, hw, 0);
    float* offb = alloc(12 * hw);
    gemm(L0.so_w, q, L0.so_b, nullptr, offb, 12, hw, d, H, W, 1, d, 1, ACT_NONE);
    float* swb = alloc(16 * hw);
    gemm(L0.sw_w, q, L0.sw_b, nullptr, swb, 16, hw, d, H, W, 1, d, 1, ACT_NONE);
    float* sampled = alloc(512 * hw); float* rpn = alloc(3 * 4 * hw);
    sample_kernel<<<dim3(hw, 4), 128, 0, stream>>>(offb, swb, ref25, l2i,
                                                   f0, f1, f2, f3, sampled, rpn, hw, 4.0f);
    float* ph = alloc(256 * 4 * hw);
    gemm(pw1, rpn, pb1, nullptr, ph, 256, 4 * hw, 3, 4 * hw, 1, 1, 1, 256, ACT_RELU);
    float* po = alloc(128 * 4 * hw);
    gemm(pw2, ph, pb2, nullptr, po, 128, 4 * hw, 256, 4 * hw, 1, 1, 1, 128, ACT_NONE);
    add_pos_k<<<ew(512 * hw), 256, 0, stream>>>(sampled, po, hw);
    float* m1 = alloc(2048 * hw);
    gemm(L0.mw0, sampled, L0.mb0, nullptr, m1, 2048, hw, 512, H, W, 1, 512, 1, ACT_GELU);
    float* m2 = alloc(2048 * hw);
    gemm(L0.mw1, m1, L0.mb1, nullptr, m2, 2048, hw, 2048, H, W, 1, 2048, 1, ACT_GELU);
    float* q2 = alloc(d * hw);
    gemm(L0.mw2, m2, L0.mb2, q, q2, d, hw, 2048, H, W, 1, 2048, 1, ACT_NONE);
    norm(q2, nullptr, nullptr, d, hw, 0);
    gemm(L0.ow0, q2, L0.ob0, nullptr, t1, d, hw, d, H, W, 1, d, 1, ACT_GELU);
    gemm(L0.ow1, t1, L0.ob1, nullptr, t2, d, hw, d, H, W, 3, d * 9, 1, ACT_GELU);
    gemm(L0.ow2, t2, L0.ob2, q2, q3, d, hw, d, H, W, 1, d, 1, ACT_NONE);
    norm(q3, nullptr, nullptr, d, hw, 0);
  }

  // ================= Decoder block: (512,25,25) -> (128,100,100) =================
  {
    off = mark;
    float* xr = alloc(512 * 10000);
    resize_bilinear_k<<<ew(512 * 10000), 256, 0, stream>>>(q3, xr, 512, 25, 25, 100, 100, 0);
    float* y1 = alloc(128 * 10000);
    gemm(dc1, xr, nullptr, nullptr, y1, 128, 10000, 512, 100, 100, 3, 512 * 9, 1, ACT_NONE);
    norm(y1, bn1_g, bn1_b, 128, 10000, 1);
    float* y2 = alloc(128 * 10000);
    gemm(dc2, y1, nullptr, nullptr, y2, 128, 10000, 128, 100, 100, 3, 128 * 9, 1, ACT_NONE);
    norm(y2, bn2_g, bn2_b, 128, 10000, 0);
    float* u25 = alloc(128 * 625);
    gemm(up_w, q3, up_b, nullptr, u25, 128, 625, 512, 25, 25, 1, 512, 1, ACT_NONE);
    float* up = alloc(128 * 10000);
    resize_nearest_k<<<ew(128 * 10000), 256, 0, stream>>>(u25, up, 128, 25, 25, 100, 100);
    add_relu_k<<<ew(128 * 10000), 256, 0, stream>>>(y2, up, bev2, 128 * 10000);
  }

  // ================= Layer 2: d=128, 100x100, scale=1 =================
  {
    off = mark;
    const int hw = 10000, H = 100, W = 100, d = 128;
    float* t1 = alloc(d * hw); float* t2 = alloc(d * hw); float* q = alloc(d * hw);
    gemm(L1.iw0, bev2, L1.ib0, nullptr, t1, d, hw, d, H, W, 1, d, 1, ACT_GELU);
    gemm(L1.iw1, t1,   L1.ib1, nullptr, t2, d, hw, d, H, W, 3, d * 9, 1, ACT_GELU);
    gemm(L1.iw2, t2,   L1.ib2, bev2,    q,  d, hw, d, H, W, 1, d, 1, ACT_NONE);
    norm(q, nullptr, nullptr, d, hw, 0);
    float* offb = alloc(12 * hw);
    gemm(L1.so_w, q, L1.so_b, nullptr, offb, 12, hw, d, H, W, 1, d, 1, ACT_NONE);
    float* swb = alloc(16 * hw);
    gemm(L1.sw_w, q, L1.sw_b, nullptr, swb, 16, hw, d, H, W, 1, d, 1, ACT_NONE);
    float* sampled = alloc(512 * hw); float* rpn = alloc(3 * 4 * hw);
    sample_kernel<<<dim3(hw, 4), 128, 0, stream>>>(offb, swb, bevp, l2i,
                                                   f0, f1, f2, f3, sampled, rpn, hw, 1.0f);
    float* ph = alloc(256 * 4 * hw);
    gemm(pw1, rpn, pb1, nullptr, ph, 256, 4 * hw, 3, 4 * hw, 1, 1, 1, 256, ACT_RELU);
    float* po = alloc(128 * 4 * hw);
    gemm(pw2, ph, pb2, nullptr, po, 128, 4 * hw, 256, 4 * hw, 1, 1, 1, 128, ACT_NONE);
    add_pos_k<<<ew(512 * hw), 256, 0, stream>>>(sampled, po, hw);
    float* m1 = alloc(512 * hw);
    gemm(L1.mw0, sampled, L1.mb0, nullptr, m1, 512, hw, 512, H, W, 1, 512, 1, ACT_GELU);
    float* m2 = alloc(512 * hw);
    gemm(L1.mw1, m1, L1.mb1, nullptr, m2, 512, hw, 512, H, W, 1, 512, 1, ACT_GELU);
    float* q2 = alloc(d * hw);
    gemm(L1.mw2, m2, L1.mb2, q, q2, d, hw, 512, H, W, 1, 512, 1, ACT_NONE);
    norm(q2, nullptr, nullptr, d, hw, 0);
    gemm(L1.ow0, q2, L1.ob0, nullptr, t1, d, hw, d, H, W, 1, d, 1, ACT_GELU);
    gemm(L1.ow1, t1, L1.ob1, nullptr, t2, d, hw, d, H, W, 3, d * 9, 1, ACT_GELU);
    float* outp = (float*)d_out;
    gemm(L1.ow2, t2, L1.ob2, q2, outp, d, hw, d, H, W, 1, d, 1, ACT_NONE);
    norm(outp, nullptr, nullptr, d, hw, 0);
  }
}
